// USEF_attentionblock_69690139345390
// MI455X (gfx1250) — compile-verified, matrix operand order checked by reading, not run on hardware
//
#include <hip/hip_runtime.h>
#include <hip/hip_bf16.h>
#include <math.h>

typedef __bf16 v16bf __attribute__((ext_vector_type(16)));
typedef float  v8f   __attribute__((ext_vector_type(8)));

#define H_   4
#define E_   8
#define D_   32
#define C_   128
#define F_   65
#define T_   2000
#define S_   2000
#define SPAD 2016          // S padded to multiple of 32
#define KQ   544           // E*F = 520 padded to 17*32
#define NV   2080          // D*F
#define EPS_ 1e-5f
#define TPW  13            // O-accumulator N-tiles per wave (10 waves * 13 * 16 = 2080)

static __device__ __forceinline__ unsigned short f2bf(float x) {
    unsigned int u = __float_as_uint(x);
    u += 0x7FFFu + ((u >> 16) & 1u);          // round-to-nearest-even
    return (unsigned short)(u >> 16);
}

// ---------------------------------------------------------------------------
// zero-fill bf16 workspace (pads must be exactly 0)
// ---------------------------------------------------------------------------
__global__ void zero_u4_kernel(uint4* __restrict__ p, size_t n4) {
    size_t i = (size_t)blockIdx.x * blockDim.x + threadIdx.x;
    size_t stride = (size_t)gridDim.x * blockDim.x;
    uint4 z; z.x = z.y = z.z = z.w = 0u;
    for (; i < n4; i += stride) p[i] = z;
}

// ---------------------------------------------------------------------------
// conv1x1 + PReLU + per-head LayerNorm over (e,F)  -> bf16 output
//   transposed==0 : out[((b*H+h)*N + n)*row_stride + e*F+f]   (qf / kf)
//   transposed==1 : out[((b*H+h)*EP*F + e*F+f)*row_stride + n] (vT)
// block = 256 threads (4 heads x 64 threads), grid = B*N
// ---------------------------------------------------------------------------
__global__ __launch_bounds__(256) void proj_ln_kernel(
    const float* __restrict__ x, int N,
    const float* __restrict__ W, const float* __restrict__ bias,
    const float* __restrict__ a,
    const float* __restrict__ gamma, const float* __restrict__ beta,
    int EP, unsigned short* __restrict__ out, int row_stride, int transposed)
{
    extern __shared__ float sm[];
    float* xl    = sm;                 // [128][66]
    float* red   = xl + C_ * 66;       // [256]
    float* red2  = red + 256;          // [256]
    float* stats = red2 + 256;         // [8] : mu/invstd per head

    const int blk = blockIdx.x;
    const int b = blk / N, n = blk % N;
    const int tid = threadIdx.x;

    // stage x[b,:,n,:] into LDS
    for (int idx = tid; idx < C_ * F_; idx += 256) {
        int c = idx / F_, f = idx - c * F_;
        xl[c * 66 + f] = x[(((size_t)b * C_ + c) * N + n) * F_ + f];
    }
    __syncthreads();

    const int h  = tid >> 6;           // head
    const int sl = tid & 63;           // slot within head
    const int nEl = EP * F_;
    const float ah = a[h];

    float s = 0.f, ss = 0.f;
    for (int idx = sl; idx < nEl; idx += 64) {
        int e = idx / F_, f = idx - e * F_;
        int o = h * EP + e;
        float acc = bias[o];
        const float* wr = W + (size_t)o * C_;
        #pragma unroll 4
        for (int c = 0; c < C_; ++c) acc = fmaf(wr[c], xl[c * 66 + f], acc);
        if (acc < 0.f) acc *= ah;
        s += acc; ss += acc * acc;
    }
    red[tid] = s; red2[tid] = ss;
    __syncthreads();
    if (tid < H_) {
        float s2 = 0.f, ss2 = 0.f;
        for (int i = 0; i < 64; ++i) { s2 += red[tid * 64 + i]; ss2 += red2[tid * 64 + i]; }
        float cnt = (float)nEl;
        float mu = s2 / cnt;
        float var = ss2 / cnt - mu * mu;
        stats[tid * 2]     = mu;
        stats[tid * 2 + 1] = rsqrtf(var + EPS_);
    }
    __syncthreads();
    const float mu = stats[h * 2], inv = stats[h * 2 + 1];
    const size_t bh = (size_t)b * H_ + h;

    for (int idx = sl; idx < nEl; idx += 64) {
        int e = idx / F_, f = idx - e * F_;
        int o = h * EP + e;
        float acc = bias[o];
        const float* wr = W + (size_t)o * C_;
        #pragma unroll 4
        for (int c = 0; c < C_; ++c) acc = fmaf(wr[c], xl[c * 66 + f], acc);
        if (acc < 0.f) acc *= ah;
        float v = (acc - mu) * inv * gamma[(size_t)o * F_ + f] + beta[(size_t)o * F_ + f];
        if (!transposed)
            out[(bh * N + n) * (size_t)row_stride + idx] = f2bf(v);
        else
            out[(bh * (size_t)nEl + idx) * (size_t)row_stride + n] = f2bf(v);
    }
}

// ---------------------------------------------------------------------------
// flash attention: one block = (b,h) x 16 query rows; 10 waves x 32 lanes
//   all 10 waves : score tile (16x32) via bf16 WMMA, K=544 split 5 ways x 2 halves
//                  (Q fragments hoisted into registers, loop-invariant)
//   all 10 waves : online-softmax rescale + 13 N-tiles of attn*V WMMA each
// ---------------------------------------------------------------------------
__global__ __launch_bounds__(320) void flash_kernel(
    const unsigned short* __restrict__ qf,
    const unsigned short* __restrict__ kf,
    const unsigned short* __restrict__ vT,
    float* __restrict__ Obuf)
{
    __shared__ float Ppart[10][16][16];         // per-wave score partials
    __shared__ unsigned short PA[16][32];       // softmaxed probs (bf16, row-major)
    __shared__ float row_alpha[16];
    __shared__ float row_l[16];

    const int blk = blockIdx.x;
    const int tt = blk % (T_ / 16);
    const int bh = blk / (T_ / 16);
    const int t0 = tt * 16;

    const int tid  = threadIdx.x;
    const int w    = tid >> 5;
    const int lane = tid & 31;
    const int nl   = lane & 15;
    const int hi   = lane >> 4;                 // 0 / 1
    const int ks   = hi ? 8 : 0;                // k sub-offset per ISA layout

    const float rscale = 0.04385290097f;        // 1/sqrt(520)

    v8f acc[TPW];
    const v8f vzero = {0.f,0.f,0.f,0.f,0.f,0.f,0.f,0.f};
    #pragma unroll
    for (int i = 0; i < TPW; ++i) acc[i] = vzero;

    float m_run = -INFINITY, l_run = 0.f;       // live in wave-0 softmax lanes

    // phase-A work split: 2 s-halves x 5 K-ranges over 10 waves
    const int nh  = w / 5;                      // which 16-wide s half
    const int qq  = w - nh * 5;
    const int kc0 = (qq * 17) / 5;              // chunk ranges: 3,3,4,3,4
    const int kc1 = ((qq + 1) * 17) / 5;
    const int nkc = kc1 - kc0;                  // wave-uniform (3 or 4)

    union Frag { uint4 u[2]; v16bf v; };

    // hoist loop-invariant Q fragments into registers (max 4 per wave)
    const unsigned short* qrow = qf + ((size_t)bh * T_ + t0 + nl) * KQ + ks;
    Frag qfr[4];
    #pragma unroll
    for (int kci = 0; kci < 4; ++kci) {
        if (kci < nkc) {
            qfr[kci].u[0] = *(const uint4*)(qrow + (kc0 + kci) * 32);
            qfr[kci].u[1] = *(const uint4*)(qrow + (kc0 + kci) * 32 + 16);
        }
    }

    for (int step = 0; step < SPAD / 32; ++step) {
        const int s0 = step * 32;

        // ---- phase A: partial score tiles (all 10 waves) -----------------
        {
            const unsigned short* krow =
                kf + ((size_t)bh * SPAD + s0 + nh * 16 + nl) * KQ + ks;
            v8f pc = vzero;
            #pragma unroll
            for (int kci = 0; kci < 4; ++kci) {
                if (kci < nkc) {                // wave-uniform branch
                    Frag bfr;
                    bfr.u[0] = *(const uint4*)(krow + (kc0 + kci) * 32);
                    bfr.u[1] = *(const uint4*)(krow + (kc0 + kci) * 32 + 16);
                    pc = __builtin_amdgcn_wmma_f32_16x16x32_bf16(
                            false, qfr[kci].v, false, bfr.v, (short)0, pc, false, false);
                }
            }
            #pragma unroll
            for (int r = 0; r < 8; ++r)
                Ppart[w][r + hi * 8][nl] = pc[r] * rscale;
        }
        __syncthreads();

        // ---- phase B: online softmax (deterministic fixed-order sums) ----
        if (tid < 16) {
            const int t = tid;
            const int valid = (s0 + 32 <= S_) ? 32 : (S_ - s0);
            float mc = -INFINITY;
            for (int j = 0; j < valid; ++j) {
                int qb = (j >= 16) ? 5 : 0, jc = j & 15;
                float v = Ppart[qb][t][jc] + Ppart[qb + 1][t][jc]
                        + Ppart[qb + 2][t][jc] + Ppart[qb + 3][t][jc]
                        + Ppart[qb + 4][t][jc];
                mc = fmaxf(mc, v);
            }
            const float mnew  = fmaxf(m_run, mc);
            const float alpha = __expf(m_run - mnew);
            float ls = 0.f;
            for (int j = 0; j < 32; ++j) {
                float p = 0.f;
                if (j < valid) {
                    int qb = (j >= 16) ? 5 : 0, jc = j & 15;
                    float v = Ppart[qb][t][jc] + Ppart[qb + 1][t][jc]
                            + Ppart[qb + 2][t][jc] + Ppart[qb + 3][t][jc]
                            + Ppart[qb + 4][t][jc];
                    p = __expf(v - mnew);
                }
                PA[t][j] = f2bf(p);
                ls += p;
            }
            l_run = l_run * alpha + ls;
            m_run = mnew;
            row_alpha[t] = alpha;
            row_l[t] = l_run;
        }
        __syncthreads();

        // ---- phase C: rescale accumulators + attn*V ----------------------
        float alr[8];
        #pragma unroll
        for (int r = 0; r < 8; ++r) alr[r] = row_alpha[r + hi * 8];

        Frag pfr;                                // A fragment from LDS
        pfr.u[0] = *(const uint4*)(&PA[nl][ks]);
        pfr.u[1] = *(const uint4*)(&PA[nl][ks + 16]);

        #pragma unroll
        for (int i = 0; i < TPW; ++i) {
            #pragma unroll
            for (int r = 0; r < 8; ++r) acc[i][r] *= alr[r];
            const int nbase = (w * TPW + i) * 16;
            const unsigned short* vrow =
                vT + ((size_t)bh * NV + nbase + nl) * SPAD + s0 + ks;
            Frag vfr;
            vfr.u[0] = *(const uint4*)(vrow);
            vfr.u[1] = *(const uint4*)(vrow + 16);
            acc[i] = __builtin_amdgcn_wmma_f32_16x16x32_bf16(
                        false, pfr.v, false, vfr.v, (short)0, acc[i], false, false);
        }
        __syncthreads();
    }

    // ---- epilogue: multiply by 1/denom, write fp32 O ---------------------
    float invl[8];
    #pragma unroll
    for (int r = 0; r < 8; ++r) invl[r] = 1.f / row_l[r + hi * 8];

    #pragma unroll
    for (int i = 0; i < TPW; ++i) {
        const int nb = (w * TPW + i) * 16 + nl;
        #pragma unroll
        for (int r = 0; r < 8; ++r) {
            const int m = r + hi * 8;
            Obuf[((size_t)bh * T_ + t0 + m) * NV + nb] = acc[i][r] * invl[r];
        }
    }
}

// ---------------------------------------------------------------------------
// output projection + PReLU(shared) + LN over (C,F) -> fp32 out
// block = 256, grid = B*T
// ---------------------------------------------------------------------------
__global__ __launch_bounds__(256) void outproj_ln_kernel(
    const float* __restrict__ Obuf,
    const float* __restrict__ Wp, const float* __restrict__ bp,
    const float* __restrict__ ap,
    const float* __restrict__ gammaO, const float* __restrict__ betaO,
    float* __restrict__ out)
{
    extern __shared__ float sm[];
    float* xl    = sm;               // [128][66]
    float* red   = xl + C_ * 66;     // [256]
    float* red2  = red + 256;        // [256]
    float* stats = red2 + 256;       // [2]

    const int blk = blockIdx.x;
    const int b = blk / T_, t = blk % T_;
    const int tid = threadIdx.x;

    for (int idx = tid; idx < C_ * F_; idx += 256) {
        int c = idx / F_, f = idx - c * F_;
        xl[c * 66 + f] =
            Obuf[(((size_t)b * H_ + (c >> 5)) * T_ + t) * NV + (c & 31) * F_ + f];
    }
    __syncthreads();

    const float aprm = ap[0];
    float s = 0.f, ss = 0.f;
    for (int idx = tid; idx < C_ * F_; idx += 256) {
        int o = idx / F_, f = idx - o * F_;
        float acc = bp[o];
        const float* wr = Wp + (size_t)o * C_;
        #pragma unroll 4
        for (int c = 0; c < C_; ++c) acc = fmaf(wr[c], xl[c * 66 + f], acc);
        if (acc < 0.f) acc *= aprm;
        s += acc; ss += acc * acc;
    }
    red[tid] = s; red2[tid] = ss;
    __syncthreads();
    if (tid == 0) {
        float s2 = 0.f, ss2 = 0.f;
        for (int i = 0; i < 256; ++i) { s2 += red[i]; ss2 += red2[i]; }
        const float cnt = (float)(C_ * F_);
        float mu = s2 / cnt;
        float var = ss2 / cnt - mu * mu;
        stats[0] = mu; stats[1] = rsqrtf(var + EPS_);
    }
    __syncthreads();
    const float mu = stats[0], inv = stats[1];
    for (int idx = tid; idx < C_ * F_; idx += 256) {
        int o = idx / F_, f = idx - o * F_;
        float acc = bp[o];
        const float* wr = Wp + (size_t)o * C_;
        #pragma unroll 4
        for (int c = 0; c < C_; ++c) acc = fmaf(wr[c], xl[c * 66 + f], acc);
        if (acc < 0.f) acc *= aprm;
        float v = (acc - mu) * inv * gammaO[idx] + betaO[idx];
        out[(((size_t)b * C_ + o) * T_ + t) * F_ + f] = v;
    }
}

// ---------------------------------------------------------------------------
extern "C" void kernel_launch(void* const* d_in, const int* in_sizes, int n_in,
                              void* d_out, int out_size, void* d_ws, size_t ws_size,
                              hipStream_t stream) {
    (void)in_sizes; (void)n_in; (void)out_size; (void)ws_size;

    const float* batch  = (const float*)d_in[0];
    const float* aux    = (const float*)d_in[1];
    const float* Wq     = (const float*)d_in[2];
    const float* bq     = (const float*)d_in[3];
    const float* Wk     = (const float*)d_in[4];
    const float* bk     = (const float*)d_in[5];
    const float* Wv     = (const float*)d_in[6];
    const float* bv     = (const float*)d_in[7];
    const float* aQ     = (const float*)d_in[8];
    const float* gammaQ = (const float*)d_in[9];
    const float* betaQ  = (const float*)d_in[10];
    const float* aK     = (const float*)d_in[11];
    const float* gammaK = (const float*)d_in[12];
    const float* betaK  = (const float*)d_in[13];
    const float* aV     = (const float*)d_in[14];
    const float* gammaV = (const float*)d_in[15];
    const float* betaV  = (const float*)d_in[16];
    const float* Wp     = (const float*)d_in[17];
    const float* bp     = (const float*)d_in[18];
    const float* ap     = (const float*)d_in[19];
    const float* gammaO = (const float*)d_in[20];
    const float* betaO  = (const float*)d_in[21];
    float* out = (float*)d_out;

    const int B = 2;
    // workspace layout (bf16 halves, then fp32 O)
    const size_t qf_n = (size_t)B * H_ * T_   * KQ;     // halfs
    const size_t kf_n = (size_t)B * H_ * SPAD * KQ;
    const size_t vT_n = (size_t)B * H_ * NV   * SPAD;
    const size_t half_total = qf_n + kf_n + vT_n;

    unsigned short* qf = (unsigned short*)d_ws;
    unsigned short* kf = qf + qf_n;
    unsigned short* vT = kf + kf_n;
    size_t obyte = ((half_total * 2 + 255) / 256) * 256;
    float* Obuf = (float*)((char*)d_ws + obyte);

    // 1) zero bf16 buffers (pads must contribute 0)
    zero_u4_kernel<<<4096, 256, 0, stream>>>((uint4*)d_ws, half_total / 8);

    // 2) projections + PReLU + per-head LN
    const size_t shm1 = (C_ * 66 + 256 + 256 + 8) * sizeof(float);
    proj_ln_kernel<<<B * T_, 256, shm1, stream>>>(
        batch, T_, Wq, bq, aQ, gammaQ, betaQ, E_, qf, KQ, 0);
    proj_ln_kernel<<<B * S_, 256, shm1, stream>>>(
        aux,   S_, Wk, bk, aK, gammaK, betaK, E_, kf, KQ, 0);
    proj_ln_kernel<<<B * S_, 256, shm1, stream>>>(
        aux,   S_, Wv, bv, aV, gammaV, betaV, D_, vT, SPAD, 1);

    // 3) flash attention (bf16 WMMA, fp32 accumulate)
    flash_kernel<<<B * H_ * (T_ / 16), 320, 0, stream>>>(qf, kf, vT, Obuf);

    // 4) output projection + PReLU + LN over (C,F)
    const size_t shm3 = (C_ * 66 + 256 + 256 + 2) * sizeof(float);
    outproj_ln_kernel<<<B * T_, 256, shm3, stream>>>(
        Obuf, Wp, bp, ap, gammaO, betaO, out);
}
